// MoE_45561013076080
// MI455X (gfx1250) — compile-verified, matrix operand order checked by reading, not run on hardware
//
#include <hip/hip_runtime.h>
#include <hip/hip_bf16.h>
#include <math.h>

#define T_TOK 4096
#define DM    1024
#define DF    2048
#define NE    8
#define LROW  (DM + 8)   // padded LDS row (bf16 elements)

typedef __bf16 bf16;
typedef __attribute__((ext_vector_type(16))) __bf16 v16bf;
typedef __attribute__((ext_vector_type(8)))  float  v8f;
typedef __attribute__((ext_vector_type(4)))  float  f32x4;

union Frag { v16bf v; f32x4 q[2]; };

__device__ __forceinline__ v8f wmma_bf16(const Frag& a, const Frag& b, v8f c) {
  return __builtin_amdgcn_wmma_f32_16x16x32_bf16(false, a.v, false, b.v,
                                                 (short)0, c, false, false);
}

// ---------------------------------------------------------------- init ----
__global__ void init_kernel(float* __restrict__ out_acc, float* __restrict__ zp,
                            int* __restrict__ cnt) {
  size_t i = (size_t)blockIdx.x * blockDim.x + threadIdx.x;
  if (i < (size_t)T_TOK * DM) out_acc[i] = 0.0f;
  if (i < 9) zp[i] = 0.0f;          // zsum + pacc[8]
  if (i < NE) cnt[i] = 0;
}

__global__ void cvt_x_kernel(const float* __restrict__ x, bf16* __restrict__ xb) {
  size_t i = (size_t)blockIdx.x * blockDim.x + threadIdx.x;
  if (i < (size_t)T_TOK * DM) xb[i] = (bf16)x[i];
}

// transpose+convert: in [R,C] f32 -> out [C,R] bf16, batched over blockIdx.z
__global__ void tconv_kernel(const float* __restrict__ in, bf16* __restrict__ out,
                             int R, int C) {
  __shared__ float tile[32][33];
  size_t base = (size_t)blockIdx.z * R * C;
  in  += base; out += base;
  int c0 = blockIdx.x * 32, r0 = blockIdx.y * 32;
  int tx = threadIdx.x, ty = threadIdx.y;   // block (32,8)
#pragma unroll
  for (int j = 0; j < 4; j++)
    tile[ty + 8 * j][tx] = in[(size_t)(r0 + ty + 8 * j) * C + c0 + tx];
  __syncthreads();
#pragma unroll
  for (int j = 0; j < 4; j++)
    out[(size_t)(c0 + ty + 8 * j) * R + r0 + tx] = (bf16)tile[tx][ty + 8 * j];
}

// -------------------------------------------------------------- router ----
__global__ void router_kernel(const float* __restrict__ x, const float* __restrict__ Wr,
                              int* __restrict__ tidx, float* __restrict__ tgate,
                              float* __restrict__ zp, int* __restrict__ cnt) {
  int t    = (int)((blockIdx.x * blockDim.x + threadIdx.x) >> 5);
  int lane = threadIdx.x & 31;
  if (t >= T_TOK) return;
  const float* xr = x + (size_t)t * DM;
  float acc[8] = {0, 0, 0, 0, 0, 0, 0, 0};
  for (int i = 0; i < DM / 32; i++) {
    int d = lane + i * 32;
    float xv = xr[d];
    const float* wr = Wr + (size_t)d * NE;
    f32x4 w0 = *(const f32x4*)wr;
    f32x4 w1 = *(const f32x4*)(wr + 4);
    acc[0] += xv * w0[0]; acc[1] += xv * w0[1]; acc[2] += xv * w0[2]; acc[3] += xv * w0[3];
    acc[4] += xv * w1[0]; acc[5] += xv * w1[1]; acc[6] += xv * w1[2]; acc[7] += xv * w1[3];
  }
#pragma unroll
  for (int e = 0; e < 8; e++)
#pragma unroll
    for (int m = 16; m >= 1; m >>= 1) acc[e] += __shfl_xor(acc[e], m, 32);
  if (lane == 0) {
    float mx = acc[0];
#pragma unroll
    for (int e = 1; e < 8; e++) mx = fmaxf(mx, acc[e]);
    float se = 0.0f;
#pragma unroll
    for (int e = 0; e < 8; e++) se += __expf(acc[e] - mx);
    float lse = __logf(se) + mx;
    int i1 = 0; float l1 = acc[0];
#pragma unroll
    for (int e = 1; e < 8; e++) if (acc[e] > l1) { l1 = acc[e]; i1 = e; }
    int i2 = -1; float l2 = -INFINITY;
#pragma unroll
    for (int e = 0; e < 8; e++) if (e != i1 && acc[e] > l2) { l2 = acc[e]; i2 = e; }
    float g1 = 1.0f / (1.0f + __expf(l2 - l1));
    tidx[t * 2] = i1;       tidx[t * 2 + 1] = i2;
    tgate[t * 2] = g1;      tgate[t * 2 + 1] = 1.0f - g1;
    atomicAdd(&zp[0], lse * lse);
#pragma unroll
    for (int e = 0; e < 8; e++) atomicAdd(&zp[1 + e], __expf(acc[e] - lse));
    atomicAdd(&cnt[i1], 1);
    atomicAdd(&cnt[i2], 1);
  }
}

__global__ void finalize_kernel(const float* __restrict__ zp, const int* __restrict__ cnt,
                                int* __restrict__ eoff, int* __restrict__ fill,
                                float* __restrict__ out_tail) {
  if (threadIdx.x == 0) {
    int o = 0; float lb = 0.0f;
    for (int e = 0; e < NE; e++) { eoff[e] = o; o += cnt[e]; fill[e] = 0; }
    float z = zp[0] / (float)T_TOK;
    for (int e = 0; e < NE; e++) {
      float f = (float)cnt[e] / (float)(T_TOK * 2);
      float p = zp[1 + e] / (float)T_TOK;
      lb += p * f;
      out_tail[e] = f;                       // tokens_per_expert
    }
    lb *= (float)NE;
    out_tail[8] = z;  out_tail[9]  = z * 0.001f;
    out_tail[10] = lb; out_tail[11] = lb * 0.1f;
  }
}

__global__ void scatter_kernel(const int* __restrict__ tidx, const float* __restrict__ tgate,
                               const int* __restrict__ eoff, int* __restrict__ fill,
                               int* __restrict__ etok, float* __restrict__ egate) {
  int t = blockIdx.x * blockDim.x + threadIdx.x;
  if (t >= T_TOK) return;
#pragma unroll
  for (int k = 0; k < 2; k++) {
    int e = tidx[t * 2 + k];
    int s = atomicAdd(&fill[e], 1);
    etok[eoff[e] + s]  = t;
    egate[eoff[e] + s] = tgate[t * 2 + k];
  }
}

// ------------------------------------------------------- expert FFN 1 ----
// grid (DF/256, T/32, NE), block 256, dyn LDS = 32*LROW*2.
// h = silu(x@W1) * (x@W3) -> hbuf (bf16). M=32, N=32 per wave; each B frag
// feeds 2 WMMAs (two M-tiles).
__global__ __launch_bounds__(256) void expert_ffn1(
    const bf16* __restrict__ xb, const bf16* __restrict__ w1t, const bf16* __restrict__ w3t,
    const int* __restrict__ etok, const int* __restrict__ eoff, const int* __restrict__ ecnt,
    bf16* __restrict__ hbuf) {
  extern __shared__ char smem[];
  bf16 (*As)[LROW] = (bf16 (*)[LROW])smem;
  int e = blockIdx.z;
  int cnt = ecnt[e];
  int m0 = blockIdx.y * 32;
  if (m0 >= cnt) return;
  int base = eoff[e];
  {
    int row = threadIdx.x >> 3;         // 0..31
    int colb = (threadIdx.x & 7) * 128; // 8 x 128 elems
    f32x4* dst = (f32x4*)(&As[row][colb]);
    int slot = m0 + row;
    if (slot < cnt) {
      const f32x4* src = (const f32x4*)(xb + (size_t)etok[base + slot] * DM + colb);
#pragma unroll
      for (int j = 0; j < 16; j++) dst[j] = src[j];
    } else {
      f32x4 z = {0.f, 0.f, 0.f, 0.f};
#pragma unroll
      for (int j = 0; j < 16; j++) dst[j] = z;
    }
  }
  __syncthreads();
  int wave = threadIdx.x >> 5, lane = threadIdx.x & 31;
  int lr = lane & 15, kg = lane >> 4;
  int nb = blockIdx.x * 256 + wave * 32;
  v8f c1[2][2], c3[2][2];
#pragma unroll
  for (int m = 0; m < 2; m++)
#pragma unroll
    for (int s = 0; s < 2; s++)
#pragma unroll
      for (int r = 0; r < 8; r++) { c1[m][s][r] = 0.0f; c3[m][s][r] = 0.0f; }

  const bf16* arow[2] = { &As[lr][kg * 8], &As[16 + lr][kg * 8] };
  const bf16* bp1[2];
  const bf16* bp3[2];
#pragma unroll
  for (int s = 0; s < 2; s++) {
    int col = nb + s * 16 + lr;
    bp1[s] = w1t + (size_t)e * DF * DM + (size_t)col * DM + kg * 8;
    bp3[s] = w3t + (size_t)e * DF * DM + (size_t)col * DM + kg * 8;
  }
#pragma unroll
  for (int kk = 0; kk < DM / 32; kk++) {
    const int k0 = kk * 32;
    Frag a[2], f1[2], f3[2];
#pragma unroll
    for (int m = 0; m < 2; m++) {
      a[m].q[0] = *(const f32x4*)(arow[m] + k0);
      a[m].q[1] = *(const f32x4*)(arow[m] + k0 + 16);
    }
#pragma unroll
    for (int s = 0; s < 2; s++) {
      f1[s].q[0] = *(const f32x4*)(bp1[s] + k0);
      f1[s].q[1] = *(const f32x4*)(bp1[s] + k0 + 16);
      f3[s].q[0] = *(const f32x4*)(bp3[s] + k0);
      f3[s].q[1] = *(const f32x4*)(bp3[s] + k0 + 16);
    }
#pragma unroll
    for (int m = 0; m < 2; m++)
#pragma unroll
      for (int s = 0; s < 2; s++) {
        c1[m][s] = wmma_bf16(a[m], f1[s], c1[m][s]);
        c3[m][s] = wmma_bf16(a[m], f3[s], c3[m][s]);
      }
  }
#pragma unroll
  for (int m = 0; m < 2; m++)
#pragma unroll
    for (int s = 0; s < 2; s++) {
      int col = nb + s * 16 + lr;
#pragma unroll
      for (int r = 0; r < 8; r++) {
        int mm = m * 16 + kg * 8 + r;
        if (m0 + mm < cnt) {
          float g = c1[m][s][r];
          float h = (g / (1.0f + __expf(-g))) * c3[m][s][r];
          hbuf[(size_t)(base + m0 + mm) * DF + col] = (bf16)h;
        }
      }
    }
}

// ------------------------------------------------------- expert FFN 2 ----
// grid (DM/256, T/32, NE), block 256, dyn LDS = 32*LROW*2 + 32*8.
// out += gate * (h @ W2). M=32, N=32 per wave; K paneled at 1024.
__global__ __launch_bounds__(256) void expert_ffn2(
    const bf16* __restrict__ hbuf, const bf16* __restrict__ w2t,
    const int* __restrict__ etok, const float* __restrict__ egate,
    const int* __restrict__ eoff, const int* __restrict__ ecnt,
    float* __restrict__ out) {
  extern __shared__ char smem[];
  bf16 (*As)[LROW] = (bf16 (*)[LROW])smem;
  float* gs = (float*)(smem + 32 * LROW * 2);
  int*   ts = (int*)(gs + 32);
  int e = blockIdx.z;
  int cnt = ecnt[e];
  int m0 = blockIdx.y * 32;
  if (m0 >= cnt) return;
  int base = eoff[e];
  if (threadIdx.x < 32) {
    int slot = m0 + threadIdx.x;
    if (slot < cnt) { gs[threadIdx.x] = egate[base + slot]; ts[threadIdx.x] = etok[base + slot]; }
    else            { gs[threadIdx.x] = 0.0f;               ts[threadIdx.x] = 0; }
  }
  int wave = threadIdx.x >> 5, lane = threadIdx.x & 31;
  int lr = lane & 15, kg = lane >> 4;
  int nb = blockIdx.x * 256 + wave * 32;
  v8f c[2][2];
#pragma unroll
  for (int m = 0; m < 2; m++)
#pragma unroll
    for (int s = 0; s < 2; s++)
#pragma unroll
      for (int r = 0; r < 8; r++) c[m][s][r] = 0.0f;

  const bf16* arow[2] = { &As[lr][kg * 8], &As[16 + lr][kg * 8] };
  const bf16* bp[2];
#pragma unroll
  for (int s = 0; s < 2; s++) {
    int col = nb + s * 16 + lr;
    bp[s] = w2t + (size_t)e * DM * DF + (size_t)col * DF + kg * 8;
  }
  for (int kp = 0; kp < DF; kp += 1024) {
    __syncthreads();
    {
      int row = threadIdx.x >> 3;
      int colb = (threadIdx.x & 7) * 128;
      f32x4* dst = (f32x4*)(&As[row][colb]);
      int slot = m0 + row;
      if (slot < cnt) {
        const f32x4* src = (const f32x4*)(hbuf + (size_t)(base + slot) * DF + kp + colb);
#pragma unroll
        for (int j = 0; j < 16; j++) dst[j] = src[j];
      } else {
        f32x4 z = {0.f, 0.f, 0.f, 0.f};
#pragma unroll
        for (int j = 0; j < 16; j++) dst[j] = z;
      }
    }
    __syncthreads();
#pragma unroll
    for (int kk = 0; kk < 32; kk++) {
      const int k0 = kk * 32;
      Frag a[2], fb[2];
#pragma unroll
      for (int m = 0; m < 2; m++) {
        a[m].q[0] = *(const f32x4*)(arow[m] + k0);
        a[m].q[1] = *(const f32x4*)(arow[m] + k0 + 16);
      }
#pragma unroll
      for (int s = 0; s < 2; s++) {
        fb[s].q[0] = *(const f32x4*)(bp[s] + kp + k0);
        fb[s].q[1] = *(const f32x4*)(bp[s] + kp + k0 + 16);
      }
#pragma unroll
      for (int m = 0; m < 2; m++)
#pragma unroll
        for (int s = 0; s < 2; s++) c[m][s] = wmma_bf16(a[m], fb[s], c[m][s]);
    }
  }
#pragma unroll
  for (int m = 0; m < 2; m++)
#pragma unroll
    for (int s = 0; s < 2; s++) {
      int col = nb + s * 16 + lr;
#pragma unroll
      for (int r = 0; r < 8; r++) {
        int mm = m * 16 + kg * 8 + r;
        if (m0 + mm < cnt)
          atomicAdd(&out[(size_t)ts[mm] * DM + col], c[m][s][r] * gs[mm]);
      }
    }
}

// ------------------------------------------------------- shared FFN 1 ----
// grid (DF/256, T/32), block 256, dyn LDS = 32*LROW*2.
__global__ __launch_bounds__(256) void shared_ffn1(
    const float* __restrict__ moe, const bf16* __restrict__ ws1t,
    const bf16* __restrict__ ws3t, bf16* __restrict__ hsbuf) {
  extern __shared__ char smem[];
  bf16 (*As)[LROW] = (bf16 (*)[LROW])smem;
  int m0 = blockIdx.y * 32;
  {
    int row = threadIdx.x >> 3;
    int colb = (threadIdx.x & 7) * 128;
    const f32x4* src = (const f32x4*)(moe + (size_t)(m0 + row) * DM + colb);
    bf16* dst = &As[row][colb];
#pragma unroll
    for (int j = 0; j < 32; j++) {
      f32x4 v = src[j];
      dst[4 * j + 0] = (bf16)v[0]; dst[4 * j + 1] = (bf16)v[1];
      dst[4 * j + 2] = (bf16)v[2]; dst[4 * j + 3] = (bf16)v[3];
    }
  }
  __syncthreads();
  int wave = threadIdx.x >> 5, lane = threadIdx.x & 31;
  int lr = lane & 15, kg = lane >> 4;
  int nb = blockIdx.x * 256 + wave * 32;
  v8f c1[2][2], c3[2][2];
#pragma unroll
  for (int m = 0; m < 2; m++)
#pragma unroll
    for (int s = 0; s < 2; s++)
#pragma unroll
      for (int r = 0; r < 8; r++) { c1[m][s][r] = 0.0f; c3[m][s][r] = 0.0f; }

  const bf16* arow[2] = { &As[lr][kg * 8], &As[16 + lr][kg * 8] };
  const bf16* bp1[2];
  const bf16* bp3[2];
#pragma unroll
  for (int s = 0; s < 2; s++) {
    int col = nb + s * 16 + lr;
    bp1[s] = ws1t + (size_t)col * DM + kg * 8;
    bp3[s] = ws3t + (size_t)col * DM + kg * 8;
  }
#pragma unroll
  for (int kk = 0; kk < DM / 32; kk++) {
    const int k0 = kk * 32;
    Frag a[2], f1[2], f3[2];
#pragma unroll
    for (int m = 0; m < 2; m++) {
      a[m].q[0] = *(const f32x4*)(arow[m] + k0);
      a[m].q[1] = *(const f32x4*)(arow[m] + k0 + 16);
    }
#pragma unroll
    for (int s = 0; s < 2; s++) {
      f1[s].q[0] = *(const f32x4*)(bp1[s] + k0);
      f1[s].q[1] = *(const f32x4*)(bp1[s] + k0 + 16);
      f3[s].q[0] = *(const f32x4*)(bp3[s] + k0);
      f3[s].q[1] = *(const f32x4*)(bp3[s] + k0 + 16);
    }
#pragma unroll
    for (int m = 0; m < 2; m++)
#pragma unroll
      for (int s = 0; s < 2; s++) {
        c1[m][s] = wmma_bf16(a[m], f1[s], c1[m][s]);
        c3[m][s] = wmma_bf16(a[m], f3[s], c3[m][s]);
      }
  }
#pragma unroll
  for (int m = 0; m < 2; m++)
#pragma unroll
    for (int s = 0; s < 2; s++) {
      int col = nb + s * 16 + lr;
#pragma unroll
      for (int r = 0; r < 8; r++) {
        int mm = m * 16 + kg * 8 + r;
        float g = c1[m][s][r];
        float h = (g / (1.0f + __expf(-g))) * c3[m][s][r];
        hsbuf[(size_t)(m0 + mm) * DF + col] = (bf16)h;
      }
    }
}

// ------------------------------------------------------- shared FFN 2 ----
// grid (DM/256, T/32), block 256, dyn LDS = 32*LROW*2. out = hs @ Ws2 (f32)
__global__ __launch_bounds__(256) void shared_ffn2(
    const bf16* __restrict__ hsbuf, const bf16* __restrict__ ws2t,
    float* __restrict__ out) {
  extern __shared__ char smem[];
  bf16 (*As)[LROW] = (bf16 (*)[LROW])smem;
  int m0 = blockIdx.y * 32;
  int wave = threadIdx.x >> 5, lane = threadIdx.x & 31;
  int lr = lane & 15, kg = lane >> 4;
  int nb = blockIdx.x * 256 + wave * 32;
  v8f c[2][2];
#pragma unroll
  for (int m = 0; m < 2; m++)
#pragma unroll
    for (int s = 0; s < 2; s++)
#pragma unroll
      for (int r = 0; r < 8; r++) c[m][s][r] = 0.0f;

  const bf16* arow[2] = { &As[lr][kg * 8], &As[16 + lr][kg * 8] };
  const bf16* bp[2];
#pragma unroll
  for (int s = 0; s < 2; s++) {
    int col = nb + s * 16 + lr;
    bp[s] = ws2t + (size_t)col * DF + kg * 8;
  }
  for (int kp = 0; kp < DF; kp += 1024) {
    __syncthreads();
    {
      int row = threadIdx.x >> 3;
      int colb = (threadIdx.x & 7) * 128;
      const f32x4* src = (const f32x4*)(hsbuf + (size_t)(m0 + row) * DF + kp + colb);
      f32x4* dst = (f32x4*)(&As[row][colb]);
#pragma unroll
      for (int j = 0; j < 16; j++) dst[j] = src[j];
    }
    __syncthreads();
#pragma unroll
    for (int kk = 0; kk < 32; kk++) {
      const int k0 = kk * 32;
      Frag a[2], fb[2];
#pragma unroll
      for (int m = 0; m < 2; m++) {
        a[m].q[0] = *(const f32x4*)(arow[m] + k0);
        a[m].q[1] = *(const f32x4*)(arow[m] + k0 + 16);
      }
#pragma unroll
      for (int s = 0; s < 2; s++) {
        fb[s].q[0] = *(const f32x4*)(bp[s] + kp + k0);
        fb[s].q[1] = *(const f32x4*)(bp[s] + kp + k0 + 16);
      }
#pragma unroll
      for (int m = 0; m < 2; m++)
#pragma unroll
        for (int s = 0; s < 2; s++) c[m][s] = wmma_bf16(a[m], fb[s], c[m][s]);
    }
  }
#pragma unroll
  for (int m = 0; m < 2; m++)
#pragma unroll
    for (int s = 0; s < 2; s++) {
      int col = nb + s * 16 + lr;
#pragma unroll
      for (int r = 0; r < 8; r++) {
        int mm = m * 16 + kg * 8 + r;
        out[(size_t)(m0 + mm) * DM + col] = c[m][s][r];
      }
    }
}

// ---------------------------------------------------------------- host ----
extern "C" void kernel_launch(void* const* d_in, const int* in_sizes, int n_in,
                              void* d_out, int out_size, void* d_ws, size_t ws_size,
                              hipStream_t stream) {
  (void)in_sizes; (void)n_in; (void)out_size; (void)ws_size;
  const float* x   = (const float*)d_in[0];
  const float* Wr  = (const float*)d_in[1];
  const float* W1  = (const float*)d_in[2];
  const float* W2  = (const float*)d_in[3];
  const float* W3  = (const float*)d_in[4];
  const float* Ws1 = (const float*)d_in[5];
  const float* Ws2 = (const float*)d_in[6];
  const float* Ws3 = (const float*)d_in[7];
  float* out = (float*)d_out;

  char* ws = (char*)d_ws;
  size_t off = 0;
  auto take = [&](size_t nbytes) -> void* {
    void* p = (void*)(ws + off);
    off += (nbytes + 255) & ~(size_t)255;
    return p;
  };
  bf16*  xb    = (bf16*) take((size_t)T_TOK * DM * 2);
  bf16*  w1t   = (bf16*) take((size_t)NE * DF * DM * 2);
  bf16*  w3t   = (bf16*) take((size_t)NE * DF * DM * 2);
  bf16*  w2t   = (bf16*) take((size_t)NE * DM * DF * 2);
  bf16*  hbuf  = (bf16*) take((size_t)2 * T_TOK * DF * 2);
  bf16*  hsbuf = (bf16*) take((size_t)T_TOK * DF * 2);
  bf16*  ws1t  = (bf16*) take((size_t)DF * DM * 2);
  bf16*  ws3t  = (bf16*) take((size_t)DF * DM * 2);
  bf16*  ws2t  = (bf16*) take((size_t)DM * DF * 2);
  int*   tidx  = (int*)  take((size_t)T_TOK * 2 * 4);
  float* tgate = (float*)take((size_t)T_TOK * 2 * 4);
  int*   etok  = (int*)  take((size_t)2 * T_TOK * 4);
  float* egate = (float*)take((size_t)2 * T_TOK * 4);
  float* zp    = (float*)take(9 * 4);   // zsum + pacc[8]
  int*   cnt   = (int*)  take(NE * 4);
  int*   eoff  = (int*)  take(NE * 4);
  int*   fill  = (int*)  take(NE * 4);

  const size_t ND = (size_t)T_TOK * DM;
  int blks = (int)((ND + 255) / 256);

  init_kernel<<<blks, 256, 0, stream>>>(out, zp, cnt);
  cvt_x_kernel<<<blks, 256, 0, stream>>>(x, xb);

  dim3 tb(32, 8);
  tconv_kernel<<<dim3(DF / 32, DM / 32, NE), tb, 0, stream>>>(W1,  w1t,  DM, DF);
  tconv_kernel<<<dim3(DM / 32, DF / 32, NE), tb, 0, stream>>>(W2,  w2t,  DF, DM);
  tconv_kernel<<<dim3(DF / 32, DM / 32, NE), tb, 0, stream>>>(W3,  w3t,  DM, DF);
  tconv_kernel<<<dim3(DF / 32, DM / 32, 1),  tb, 0, stream>>>(Ws1, ws1t, DM, DF);
  tconv_kernel<<<dim3(DM / 32, DF / 32, 1),  tb, 0, stream>>>(Ws2, ws2t, DF, DM);
  tconv_kernel<<<dim3(DF / 32, DM / 32, 1),  tb, 0, stream>>>(Ws3, ws3t, DM, DF);

  router_kernel<<<T_TOK / 8, 256, 0, stream>>>(x, Wr, tidx, tgate, zp, cnt);
  finalize_kernel<<<1, 32, 0, stream>>>(zp, cnt, eoff, fill, out + ND);
  scatter_kernel<<<T_TOK / 256, 256, 0, stream>>>(tidx, tgate, eoff, fill, etok, egate);

  const size_t lds_a  = (size_t)32 * LROW * 2;           // 66 KB A-panel
  const size_t lds_a2 = lds_a + 32 * 4 + 32 * 4;         // + gates/tokens

  expert_ffn1<<<dim3(DF / 256, T_TOK / 32, NE), 256, lds_a, stream>>>(
      xb, w1t, w3t, etok, eoff, cnt, hbuf);
  expert_ffn2<<<dim3(DM / 256, T_TOK / 32, NE), 256, lds_a2, stream>>>(
      hbuf, w2t, etok, egate, eoff, cnt, out);

  shared_ffn1<<<dim3(DF / 256, T_TOK / 32), 256, lds_a, stream>>>(out, ws1t, ws3t, hsbuf);
  shared_ffn2<<<dim3(DM / 256, T_TOK / 32), 256, lds_a, stream>>>(hsbuf, ws2t, out);
}
